// SA2_89721866813656
// MI455X (gfx1250) — matrix operand order repaired and checked by reading, not compiled, not run on hardware
//
#include <hip/hip_runtime.h>
#include <hip/hip_bf16.h>

#define B_  32
#define C_  1024
#define N_  1024
#define C8_ 128

typedef __attribute__((ext_vector_type(16))) __bf16 v16bf;
typedef __attribute__((ext_vector_type(8)))  float  v8f;

union Frag { v16bf v; uint4 q[2]; __bf16 e[16]; };
union Acc  { v8f v; float f[8]; };

__device__ __forceinline__ v8f wmma_bf16(const Frag& a, const Frag& b, v8f c) {
  return __builtin_amdgcn_wmma_f32_16x16x32_bf16(false, a.v, false, b.v, (short)0, c,
                                                 false, false);
}

// ---- WMMA fragment loaders (pointer = lane base + folded immediate) ---------
// A 16x32 (MxK), 16-bit: lanes 0-15 row M=lane, K {0..15};
//                        lanes 16-31 row M=lane-16, K {16..31}.
__device__ __forceinline__ Frag ldA(const __bf16* p) {
  Frag f;
  f.q[0] = *(const uint4*)p;
  f.q[1] = *(const uint4*)(p + 8);
  return f;
}
__device__ __forceinline__ Frag ldA(const float* p) {
  Frag f;
#pragma unroll
  for (int i = 0; i < 8; ++i) f.e[i] = (__bf16)p[i];
#pragma unroll
  for (int i = 0; i < 8; ++i) f.e[8 + i] = (__bf16)p[8 + i];
  return f;
}
// B 32x16 (KxN) from Bt row-major (N x K): lanes 0-15 col N=lane, K {0..15};
//                                          lanes 16-31 col N=lane-16, K {16..31}.
__device__ __forceinline__ Frag ldB(const __bf16* p) {
  Frag f;
  f.q[0] = *(const uint4*)p;
  f.q[1] = *(const uint4*)(p + 8);
  return f;
}
__device__ __forceinline__ Frag ldB(const float* p) {
  Frag f;
#pragma unroll
  for (int i = 0; i < 16; ++i) f.e[i] = (__bf16)p[i];
  return f;
}

// Per-wave 64x32 tile (4x2 accumulators); block = 8 waves = 128(M) x 128(N).
// Apl/Bpl are per-lane base pointers (tile row/col + lane offsets already added);
// mi*16*LD / nj*16*LD fold into the global_load immediate offset after unroll.
template <int LDA, int LDB, typename TA, typename TB>
__device__ __forceinline__ void gemm_core(const TA* __restrict__ Apl,
                                          const TB* __restrict__ Bpl,
                                          int K, Acc (&acc)[4][2]) {
  for (int k0 = 0; k0 < K; k0 += 32) {
    Frag fa[4], fb[2];
#pragma unroll
    for (int mi = 0; mi < 4; ++mi) fa[mi] = ldA(Apl + mi * 16 * LDA);
#pragma unroll
    for (int nj = 0; nj < 2; ++nj) fb[nj] = ldB(Bpl + nj * 16 * LDB);
#pragma unroll
    for (int mi = 0; mi < 4; ++mi)
#pragma unroll
      for (int nj = 0; nj < 2; ++nj)
        acc[mi][nj].v = wmma_bf16(fa[mi], fb[nj], acc[mi][nj].v);
    Apl += 32;
    Bpl += 32;
  }
}

#define WAVE_TILE_SETUP()                                              \
  const int lane = threadIdx.x & 31, wave = threadIdx.x >> 5;          \
  const int wm = (wave & 1) << 6, wn = (wave >> 1) << 5;               \
  const int b = blockIdx.z;                                            \
  const int col = lane & 15, mo = (lane >> 4) << 3;                    \
  const int laneA = (lane & 15), laneAk = (lane >> 4) << 4;            \
  const int laneB = (lane & 15), laneBk = (lane >> 4) << 4;

// C/D 16x16 f32 layout: element r of acc at lane L -> row M = r + 8*(L>=16), col = L&15.

// ---- 0) elementwise f32 -> bf16 (weights) -----------------------------------
__global__ void SA2_cvt_bf16(const float* __restrict__ src, __bf16* __restrict__ dst, int n) {
  const int i = blockIdx.x * 256 + threadIdx.x;
  if (i < n) dst[i] = (__bf16)src[i];
}

// ---- 1) transpose + f32->bf16 convert: (B,C,N) -> (B,N,C) -------------------
__global__ void SA2_transpose_cvt(const float* __restrict__ src, __bf16* __restrict__ dst) {
  __shared__ float tile[32][33];
  const int b = blockIdx.z, c0 = blockIdx.y << 5, n0 = blockIdx.x << 5;
  const int tx = threadIdx.x & 31, ty = threadIdx.x >> 5;  // 256 thr: ty 0..7
  const float* s = src + (size_t)b * C_ * N_;
#pragma unroll
  for (int i = 0; i < 32; i += 8)
    tile[ty + i][tx] = s[(size_t)(c0 + ty + i) * N_ + n0 + tx];
  __syncthreads();
  __bf16* d = dst + (size_t)b * N_ * C_;
#pragma unroll
  for (int i = 0; i < 32; i += 8)
    d[(size_t)(n0 + ty + i) * C_ + c0 + tx] = (__bf16)tile[tx][ty + i];
}

// ---- 2) q/k projection: qT[b,n,o] = bias[o] + sum_c W[o,c] * Xt[b,n,c] ------
__global__ void __launch_bounds__(256)
SA2_proj_qk(const __bf16* __restrict__ W, const float* __restrict__ bias,
            const __bf16* __restrict__ Xt, __bf16* __restrict__ qT) {
  WAVE_TILE_SETUP();
  const int Mb = wm;                        // C8 = 128 = one block tile in M
  const int Nb = (blockIdx.x << 7) + wn;
  const __bf16* Apl = W + (size_t)(Mb + laneA) * C_ + laneAk;
  const __bf16* Bpl = Xt + (size_t)b * N_ * C_ + (size_t)(Nb + laneB) * C_ + laneBk;
  Acc acc[4][2] = {};
  gemm_core<C_, C_>(Apl, Bpl, C_, acc);
  __bf16* out = qT + (size_t)b * N_ * C8_;
#pragma unroll
  for (int mi = 0; mi < 4; ++mi)
#pragma unroll
    for (int nj = 0; nj < 2; ++nj) {
      const int ob = Mb + mi * 16 + mo, n = Nb + nj * 16 + col;
#pragma unroll
      for (int r = 0; r < 8; ++r)       // transposed store -> o-contiguous rows
        out[(size_t)n * C8_ + ob + r] = (__bf16)(acc[mi][nj].f[r] + bias[ob + r]);
    }
}

// ---- 3) energy_c[b,n,m] = sum_o qT[n,o] * kT[m,o] (K = 128) -----------------
__global__ void __launch_bounds__(256)
SA2_energy_c(const __bf16* __restrict__ qT, const __bf16* __restrict__ kT,
             float* __restrict__ E) {
  WAVE_TILE_SETUP();
  const int Mb = (blockIdx.y << 7) + wm, Nb = (blockIdx.x << 7) + wn;
  const __bf16* Apl = qT + (size_t)b * N_ * C8_ + (size_t)(Mb + laneA) * C8_ + laneAk;
  const __bf16* Bpl = kT + (size_t)b * N_ * C8_ + (size_t)(Nb + laneB) * C8_ + laneBk;
  Acc acc[4][2] = {};
  gemm_core<C8_, C8_>(Apl, Bpl, C8_, acc);
  float* out = E + (size_t)b * N_ * N_;
#pragma unroll
  for (int mi = 0; mi < 4; ++mi)
#pragma unroll
    for (int nj = 0; nj < 2; ++nj)
#pragma unroll
      for (int r = 0; r < 8; ++r)
        out[(size_t)(Mb + mi * 16 + mo + r) * N_ + Nb + nj * 16 + col] = acc[mi][nj].f[r];
}

// ---- 5) energy_s[b,c,d] = sum_n x[c,n] * y[d,n] (f32 sources, cvt on load) --
__global__ void __launch_bounds__(256)
SA2_energy_s(const float* __restrict__ x, const float* __restrict__ y,
             float* __restrict__ E) {
  WAVE_TILE_SETUP();
  const int Mb = (blockIdx.y << 7) + wm, Nb = (blockIdx.x << 7) + wn;
  const float* Apl = x + (size_t)b * C_ * N_ + (size_t)(Mb + laneA) * N_ + laneAk;
  const float* Bpl = y + (size_t)b * C_ * N_ + (size_t)(Nb + laneB) * N_ + laneBk;
  Acc acc[4][2] = {};
  gemm_core<N_, N_>(Apl, Bpl, N_, acc);
  float* out = E + (size_t)b * N_ * N_;
#pragma unroll
  for (int mi = 0; mi < 4; ++mi)
#pragma unroll
    for (int nj = 0; nj < 2; ++nj)
#pragma unroll
      for (int r = 0; r < 8; ++r)
        out[(size_t)(Mb + mi * 16 + mo + r) * N_ + Nb + nj * 16 + col] = acc[mi][nj].f[r];
}

// ---- 7) v[b,c,n] = bv[c] + sum_d Wv[c,d] * Zt[b,n,d] ------------------------
__global__ void __launch_bounds__(256)
SA2_vproj(const __bf16* __restrict__ Wv, const float* __restrict__ bv,
          const __bf16* __restrict__ Zt, __bf16* __restrict__ V) {
  WAVE_TILE_SETUP();
  const int Mb = (blockIdx.y << 7) + wm, Nb = (blockIdx.x << 7) + wn;
  const __bf16* Apl = Wv + (size_t)(Mb + laneA) * C_ + laneAk;
  const __bf16* Bpl = Zt + (size_t)b * N_ * C_ + (size_t)(Nb + laneB) * C_ + laneBk;
  Acc acc[4][2] = {};
  gemm_core<C_, C_>(Apl, Bpl, C_, acc);
  __bf16* out = V + (size_t)b * C_ * N_;
#pragma unroll
  for (int mi = 0; mi < 4; ++mi)
#pragma unroll
    for (int nj = 0; nj < 2; ++nj)
#pragma unroll
      for (int r = 0; r < 8; ++r) {
        const int c = Mb + mi * 16 + mo + r;
        out[(size_t)c * N_ + Nb + nj * 16 + col] = (__bf16)(acc[mi][nj].f[r] + bv[c]);
      }
}

// ---- 8) out[b,c,m] = z[b,c,m] + sum_n V[c,n] * att[m,n] ---------------------
__global__ void __launch_bounds__(256)
SA2_out(const __bf16* __restrict__ V, const __bf16* __restrict__ att,
        const float* __restrict__ z, float* __restrict__ out) {
  WAVE_TILE_SETUP();
  const int Mb = (blockIdx.y << 7) + wm, Nb = (blockIdx.x << 7) + wn;
  const __bf16* Apl = V + (size_t)b * C_ * N_ + (size_t)(Mb + laneA) * N_ + laneAk;
  const __bf16* Bpl = att + (size_t)b * N_ * N_ + (size_t)(Nb + laneB) * N_ + laneBk;
  Acc acc[4][2] = {};
  gemm_core<N_, N_>(Apl, Bpl, N_, acc);
  const float* zb = z + (size_t)b * C_ * N_;
  float* ob = out + (size_t)b * C_ * N_;
#pragma unroll
  for (int mi = 0; mi < 4; ++mi)
#pragma unroll
    for (int nj = 0; nj < 2; ++nj)
#pragma unroll
      for (int r = 0; r < 8; ++r) {
        const size_t idx = (size_t)(Mb + mi * 16 + mo + r) * N_ + Nb + nj * 16 + col;
        ob[idx] = zb[idx] + acc[mi][nj].f[r];
      }
}

// ---- 4) softmax over energy_c rows -> attn_c bf16 ---------------------------
__global__ void SA2_softmax_c(const float* __restrict__ E, __bf16* __restrict__ Ac) {
  const int row = blockIdx.x, t = threadIdx.x;
  const float* e = E + (size_t)row * N_;
  __shared__ float red[256];
  float v[4]; float mx = -3.4e38f;
#pragma unroll
  for (int i = 0; i < 4; ++i) { v[i] = e[t + (i << 8)]; mx = fmaxf(mx, v[i]); }
  red[t] = mx; __syncthreads();
  for (int s = 128; s > 0; s >>= 1) { if (t < s) red[t] = fmaxf(red[t], red[t + s]); __syncthreads(); }
  mx = red[0]; __syncthreads();
  float sum = 0.f;
#pragma unroll
  for (int i = 0; i < 4; ++i) { v[i] = __expf(v[i] - mx); sum += v[i]; }
  red[t] = sum; __syncthreads();
  for (int s = 128; s > 0; s >>= 1) { if (t < s) red[t] += red[t + s]; __syncthreads(); }
  const float inv = 1.f / red[0];
  __bf16* o = Ac + (size_t)row * N_;
#pragma unroll
  for (int i = 0; i < 4; ++i) o[t + (i << 8)] = (__bf16)(v[i] * inv);
}

// ---- 6) fused: attn_s = softmax(rowmax - energy_s); att = softmax(attn_c*attn_s)
__global__ void SA2_attn_fuse(const float* __restrict__ Es, const __bf16* __restrict__ Ac,
                              __bf16* __restrict__ att) {
  const int row = blockIdx.x, t = threadIdx.x;
  const float* e = Es + (size_t)row * N_;
  const __bf16* a = Ac + (size_t)row * N_;
  __shared__ float red[256];
  float v[4]; float mx = -3.4e38f;
#pragma unroll
  for (int i = 0; i < 4; ++i) { v[i] = e[t + (i << 8)]; mx = fmaxf(mx, v[i]); }
  red[t] = mx; __syncthreads();
  for (int s = 128; s > 0; s >>= 1) { if (t < s) red[t] = fmaxf(red[t], red[t + s]); __syncthreads(); }
  mx = red[0]; __syncthreads();
  // softmax of (mx - e): min of transformed row is 0; exp taken without a max shift.
  float sum = 0.f;
#pragma unroll
  for (int i = 0; i < 4; ++i) { v[i] = __expf(mx - v[i]); sum += v[i]; }
  red[t] = sum; __syncthreads();
  for (int s = 128; s > 0; s >>= 1) { if (t < s) red[t] += red[t + s]; __syncthreads(); }
  const float inv = 1.f / red[0]; __syncthreads();
  // combine with attn_c, then second softmax
  float u[4]; float umax = -3.4e38f;
#pragma unroll
  for (int i = 0; i < 4; ++i) { u[i] = (float)a[t + (i << 8)] * (v[i] * inv); umax = fmaxf(umax, u[i]); }
  red[t] = umax; __syncthreads();
  for (int s = 128; s > 0; s >>= 1) { if (t < s) red[t] = fmaxf(red[t], red[t + s]); __syncthreads(); }
  umax = red[0]; __syncthreads();
  float s2 = 0.f;
#pragma unroll
  for (int i = 0; i < 4; ++i) { u[i] = __expf(u[i] - umax); s2 += u[i]; }
  red[t] = s2; __syncthreads();
  for (int s = 128; s > 0; s >>= 1) { if (t < s) red[t] += red[t + s]; __syncthreads(); }
  const float inv2 = 1.f / red[0];
  __bf16* o = att + (size_t)row * N_;
#pragma unroll
  for (int i = 0; i < 4; ++i) o[t + (i << 8)] = (__bf16)(u[i] * inv2);
}

extern "C" void kernel_launch(void* const* d_in, const int* in_sizes, int n_in,
                              void* d_out, int out_size, void* d_ws, size_t ws_size,
                              hipStream_t stream) {
  (void)in_sizes; (void)n_in; (void)out_size; (void)ws_size;
  const float* z  = (const float*)d_in[0];
  const float* x  = (const float*)d_in[1];
  const float* y  = (const float*)d_in[2];
  const float* Wq = (const float*)d_in[3];
  const float* bq = (const float*)d_in[4];
  const float* Wk = (const float*)d_in[5];
  const float* bk = (const float*)d_in[6];
  const float* Wv = (const float*)d_in[7];
  const float* bv = (const float*)d_in[8];
  float* out = (float*)d_out;

  // Workspace layout (peak ~339 MB, regions reused once dead):
  char* ws = (char*)d_ws;
  const size_t SZbf = (size_t)B_ * N_ * C_ * sizeof(__bf16);   // 64 MB
  const size_t SZqk = (size_t)B_ * N_ * C8_ * sizeof(__bf16);  //  8 MB
  const size_t SZE  = (size_t)B_ * N_ * N_ * sizeof(float);    // 128 MB
  __bf16* XT  = (__bf16*)(ws);                                 // x^T bf16 (B,N,C)
  __bf16* YT  = (__bf16*)(ws + SZbf);                          // y^T bf16
  __bf16* ZT  = (__bf16*)(ws + 2 * SZbf);                      // z^T bf16
  __bf16* qT  = (__bf16*)(ws + 3 * SZbf);                      // (B,N,C8)
  __bf16* kT  = (__bf16*)(ws + 3 * SZbf + SZqk);               // (B,N,C8)
  float*  E   = (float*) (ws + 3 * SZbf + 2 * SZqk);           // (B,N,N) f32
  __bf16* Wqb = (__bf16*)(ws + 3 * SZbf + 2 * SZqk + SZE);     // weights bf16
  __bf16* Wkb = Wqb + (size_t)C8_ * C_;
  __bf16* Wvb = Wkb + (size_t)C8_ * C_;
  __bf16* Ac  = XT;          // attn_c bf16 — XT dead after projections
  __bf16* AT  = YT;          // final attention bf16 — YT dead after projections
  __bf16* V   = (__bf16*)E;  // v bf16 — E dead after fused attention

  const dim3 blk(256, 1, 1);
  const dim3 gT(N_ / 32, C_ / 32, B_);
  const dim3 gP(N_ / 128, 1, B_);
  const dim3 gG(N_ / 128, N_ / 128, B_);

  // weights -> bf16 (once)
  SA2_cvt_bf16<<<dim3((C8_ * C_) / 256), blk, 0, stream>>>(Wq, Wqb, C8_ * C_);
  SA2_cvt_bf16<<<dim3((C8_ * C_) / 256), blk, 0, stream>>>(Wk, Wkb, C8_ * C_);
  SA2_cvt_bf16<<<dim3((C_ * C_) / 256),  blk, 0, stream>>>(Wv, Wvb, C_ * C_);

  SA2_transpose_cvt<<<gT, blk, 0, stream>>>(x, XT);
  SA2_transpose_cvt<<<gT, blk, 0, stream>>>(y, YT);
  SA2_transpose_cvt<<<gT, blk, 0, stream>>>(z, ZT);

  SA2_proj_qk<<<gP, blk, 0, stream>>>(Wqb, bq, XT, qT);
  SA2_proj_qk<<<gP, blk, 0, stream>>>(Wkb, bk, YT, kT);

  SA2_energy_c<<<gG, blk, 0, stream>>>(qT, kT, E);
  SA2_softmax_c<<<dim3(B_ * N_), blk, 0, stream>>>(E, Ac);

  SA2_energy_s<<<gG, blk, 0, stream>>>(x, y, E);
  SA2_attn_fuse<<<dim3(B_ * C_), blk, 0, stream>>>(E, Ac, AT);

  SA2_vproj<<<gG, blk, 0, stream>>>(Wvb, bv, ZT, V);
  SA2_out<<<gG, blk, 0, stream>>>(V, AT, z, out);
}
